// SpatialRootLayer_16088947491592
// MI455X (gfx1250) — matrix operand order repaired and checked by reading, not compiled, not run on hardware
//
#include <hip/hip_runtime.h>
#include <hip/hip_bf16.h>

typedef __attribute__((ext_vector_type(16))) _Float16 v16h;
typedef __attribute__((ext_vector_type(8)))  _Float16 v8h;
typedef __attribute__((ext_vector_type(8)))  float    v8f;

#define Bdim 128
#define Odim 128
#define Fdim 32768   // 32*32*32

static __device__ __forceinline__ v16h cat16(v8h lo, v8h hi) {
  return __builtin_shufflevector(lo, hi, 0, 1, 2, 3, 4, 5, 6, 7,
                                          8, 9, 10, 11, 12, 13, 14, 15);
}

// ---------------------------------------------------------------------------
// Kernel 1: per-row max of x, then Aexp[b,f] = exp(x[b,f]-mx[b]) as f16.
// ---------------------------------------------------------------------------
__global__ void __launch_bounds__(256)
row_exp_x(const float* __restrict__ x, _Float16* __restrict__ Aexp,
          float* __restrict__ mx) {
  const int row = blockIdx.x;
  const int tid = threadIdx.x;
  const float* __restrict__ xr = x + (size_t)row * Fdim;
  __shared__ float red[256];

  float m = -3.4e38f;
  for (int i = tid; i < Fdim; i += 256) m = fmaxf(m, xr[i]);
  red[tid] = m;
  __syncthreads();
  for (int s = 128; s > 0; s >>= 1) {
    if (tid < s) red[tid] = fmaxf(red[tid], red[tid + s]);
    __syncthreads();
  }
  const float rm = red[0];

  _Float16* __restrict__ ar = Aexp + (size_t)row * Fdim;
  for (int i = tid; i < Fdim; i += 256)
    ar[i] = (_Float16)__expf(xr[i] - rm);
  if (tid == 0) mx[row] = rm;
}

// ---------------------------------------------------------------------------
// Kernel 2: per-row max of weight, Wexp[o,f] = exp(w-mw) as f16,
// and lsw[o] = log(sum_f exp(w-mw)).
// ---------------------------------------------------------------------------
__global__ void __launch_bounds__(256)
row_exp_w(const float* __restrict__ w, _Float16* __restrict__ Wexp,
          float* __restrict__ lsw) {
  const int row = blockIdx.x;
  const int tid = threadIdx.x;
  const float* __restrict__ wr = w + (size_t)row * Fdim;
  __shared__ float red[256];

  float m = -3.4e38f;
  for (int i = tid; i < Fdim; i += 256) m = fmaxf(m, wr[i]);
  red[tid] = m;
  __syncthreads();
  for (int s = 128; s > 0; s >>= 1) {
    if (tid < s) red[tid] = fmaxf(red[tid], red[tid + s]);
    __syncthreads();
  }
  const float rm = red[0];
  __syncthreads();   // red[] reused below

  _Float16* __restrict__ we = Wexp + (size_t)row * Fdim;
  float acc = 0.f;
  for (int i = tid; i < Fdim; i += 256) {
    const float e = __expf(wr[i] - rm);
    we[i] = (_Float16)e;
    acc += e;
  }
  red[tid] = acc;
  __syncthreads();
  for (int s = 128; s > 0; s >>= 1) {
    if (tid < s) red[tid] += red[tid + s];
    __syncthreads();
  }
  if (tid == 0) lsw[row] = __logf(red[0]);
}

// ---------------------------------------------------------------------------
// Kernel 3: WMMA GEMM  Spart[z, b, o] = sum_{f in K-quarter z} Aexp[b,f]*Wexp[o,f]
// Grid (8,8,4): 16x16 output tile per block, K split 4x over z and 8x over the
// block's eight wave32s. Two K-steps (K=64) per trip with independent
// accumulators so the two v_wmma ops can overlap and loads clause as 8x b128.
// Fragment layouts used here (wave32):
//   A (16x32 f16): lane L, M=L&15; K = 16*(L>=16) .. +15 (contiguous)
//   B (32x16 f16): lane L, N=L&15; K = 16*(L>=16) .. +15 (contiguous)
//   C/D (16x16 f32): VGPR r, lane L -> M = r + 8*(L>=16), N = L&15
// ---------------------------------------------------------------------------
__global__ void __launch_bounds__(256)
logmm_wmma(const _Float16* __restrict__ A, const _Float16* __restrict__ W,
           float* __restrict__ Spart) {
  const int m0   = blockIdx.x * 16;
  const int n0   = blockIdx.y * 16;
  const int tid  = threadIdx.x;
  const int lane = tid & 31;
  const int wave = tid >> 5;
  const int lm   = lane & 15;
  const int hi   = lane >> 4;

  const int kQuarter = Fdim >> 2;                       // 8192
  const int kChunk   = kQuarter >> 3;                   // 1024 per wave
  const int kBegin   = blockIdx.z * kQuarter + wave * kChunk;

  // Running pointers; all subsequent accesses are immediate offsets off these.
  const _Float16* pa = A + (size_t)(m0 + lm) * Fdim + kBegin + hi * 16;
  const _Float16* pb = W + (size_t)(n0 + lm) * Fdim + kBegin + hi * 16;

  v8f c0 = {}, c1 = {};
  for (int it = 0; it < kChunk; it += 64) {
    // K-step 0 fragments
    const v8h a00 = *(const v8h*)(pa);          // K 16*hi .. +7
    const v8h a01 = *(const v8h*)(pa + 8);      // K 16*hi+8 .. +15
    const v8h b00 = *(const v8h*)(pb);          // K 16*hi .. +7
    const v8h b01 = *(const v8h*)(pb + 8);      // K 16*hi+8 .. +15
    // K-step 1 fragments (+32 elements)
    const v8h a10 = *(const v8h*)(pa + 32);
    const v8h a11 = *(const v8h*)(pa + 40);
    const v8h b10 = *(const v8h*)(pb + 32);
    const v8h b11 = *(const v8h*)(pb + 40);

    __builtin_prefetch(pa + 512, 0, 3);         // near-scope global_prefetch
    __builtin_prefetch(pb + 512, 0, 3);

    // (neg_a, A, neg_b, B, c_mod, C, reuse_a, reuse_b)
    c0 = __builtin_amdgcn_wmma_f32_16x16x32_f16(
        false, cat16(a00, a01), false, cat16(b00, b01), (short)0, c0, false, false);
    c1 = __builtin_amdgcn_wmma_f32_16x16x32_f16(
        false, cat16(a10, a11), false, cat16(b10, b11), (short)0, c1, false, false);

    pa += 64;
    pb += 64;
  }

  // Deterministic cross-wave reduction of the 8 partial 16x16 tiles.
  __shared__ float acc[8][256];
#pragma unroll
  for (int r = 0; r < 8; ++r)
    acc[wave][(r + hi * 8) * 16 + lm] = c0[r] + c1[r];
  __syncthreads();

  float s = 0.f;
#pragma unroll
  for (int wv = 0; wv < 8; ++wv) s += acc[wv][tid];

  const int row = tid >> 4;
  const int col = tid & 15;
  Spart[(size_t)blockIdx.z * (Bdim * Odim) + (m0 + row) * Odim + (n0 + col)] = s;
}

// ---------------------------------------------------------------------------
// Kernel 4: out[b,o] = mx[b] + log(sum_z Spart[z,b,o]) - lsw[o]
// ---------------------------------------------------------------------------
__global__ void __launch_bounds__(256)
finalize_k(const float* __restrict__ Spart, const float* __restrict__ mx,
           const float* __restrict__ lsw, float* __restrict__ out) {
  const int i = blockIdx.x * 256 + threadIdx.x;   // 0 .. 16383
  const int N = Bdim * Odim;
  const float s = Spart[i] + Spart[N + i] + Spart[2 * N + i] + Spart[3 * N + i];
  const int b = i >> 7;
  const int o = i & 127;
  out[i] = mx[b] + __logf(s) - lsw[o];
}

// ---------------------------------------------------------------------------
extern "C" void kernel_launch(void* const* d_in, const int* in_sizes, int n_in,
                              void* d_out, int out_size, void* d_ws, size_t ws_size,
                              hipStream_t stream) {
  const float* x = (const float*)d_in[0];        // [128,32,32,32] fp32
  const float* w = (const float*)d_in[1];        // [128,32768]    fp32
  float* out = (float*)d_out;                    // [128,128]      fp32

  // Workspace carve-up
  char* ws = (char*)d_ws;
  const size_t ABYTES = (size_t)Bdim * Fdim * sizeof(_Float16);   // 8 MB
  _Float16* Aexp  = (_Float16*)ws;                 ws += ABYTES;
  _Float16* Wexp  = (_Float16*)ws;                 ws += ABYTES;
  float*    Spart = (float*)ws;                    ws += (size_t)4 * Bdim * Odim * sizeof(float);
  float*    mx    = (float*)ws;                    ws += Bdim * sizeof(float);
  float*    lsw   = (float*)ws;

  row_exp_x<<<Bdim, 256, 0, stream>>>(x, Aexp, mx);
  row_exp_w<<<Odim, 256, 0, stream>>>(w, Wexp, lsw);

  dim3 grid(Bdim / 16, Odim / 16, 4);              // (8,8,4)
  logmm_wmma<<<grid, 256, 0, stream>>>(Aexp, Wexp, Spart);

  finalize_k<<<(Bdim * Odim) / 256, 256, 0, stream>>>(Spart, mx, lsw, out);
}
